// GraphTransformer_72507637891225
// MI455X (gfx1250) — compile-verified, matrix operand order checked by reading, not run on hardware
//
#include <hip/hip_runtime.h>
#include <hip/hip_bf16.h>

// ---------------------------------------------------------------------------
// GraphTransformer on MI455X (gfx1250, wave32).
// Dense node GEMMs  -> v_wmma_f32_16x16x32_f16 (f16 in, f32 accum), packed
//                      weights staged block-wide in LDS via async global->LDS.
// Edge attention    -> 3 bandwidth-bound passes with f16 gathers + f32 atomics.
// Edge-feature proj -> recomputed per edge from LDS-staged 128x3 weight.
// ---------------------------------------------------------------------------

typedef _Float16 v16h __attribute__((ext_vector_type(16)));
typedef _Float16 v8h  __attribute__((ext_vector_type(8)));
typedef float    v8f  __attribute__((ext_vector_type(8)));
typedef int      v4i  __attribute__((ext_vector_type(4)));

union AFrag { v16h v; v8h h[2]; };

#if defined(__has_builtin)
#  if __has_builtin(__builtin_amdgcn_global_load_async_to_lds_b128)
#    define HAS_ASYNC_LDS 1
#  else
#    define HAS_ASYNC_LDS 0
#  endif
#else
#  define HAS_ASYNC_LDS 0
#endif

__device__ __forceinline__ float gelu_tanh(float x) {
    float x3 = x * x * x;
    return 0.5f * x * (1.0f + tanhf(0.7978845608028654f * (x + 0.044715f * x3)));
}

// float atomic max via sign-split int/uint trick (valid with -inf init)
__device__ __forceinline__ void atomicMaxF(float* addr, float v) {
    if (v >= 0.0f) atomicMax((int*)addr, __float_as_int(v));
    else           atomicMin((unsigned int*)addr, __float_as_uint(v));
}

// ---------------------------------------------------------------------------
// Pack one 128-col-K weight matrix (ncols x 128, row-major f32) into WMMA
// B-fragment layout: frag (kc,nt) -> 32 lanes x 16 contiguous f16 per lane.
// Lane L holds column n = nt*16+(L&15); elems 0..7 are K=kc*32+(L>>4)*8+i,
// elems 8..15 are K=kc*32+16+(L>>4)*8+(i-8)  (mirrors the A-matrix layout).
// ---------------------------------------------------------------------------
__global__ __launch_bounds__(32) void pack_w(const float* __restrict__ W,
                                             _Float16* __restrict__ out,
                                             int ntiles) {
    int frag = blockIdx.x;                 // 0 .. 4*ntiles-1
    int kc = frag / ntiles, nt = frag % ntiles;
    int lane = threadIdx.x;                // 0..31
    int n    = nt * 16 + (lane & 15);
    int half = lane >> 4;
    _Float16* dst = out + (size_t)frag * 512 + lane * 16;
#pragma unroll
    for (int i = 0; i < 16; ++i) {
        int k = kc * 32 + ((i >= 8) ? 16 : 0) + half * 8 + (i & 7);
        dst[i] = (_Float16)W[(size_t)n * 128 + k];
    }
}

__global__ __launch_bounds__(256) void conv16(const float* __restrict__ in,
                                              _Float16* __restrict__ out, size_t n) {
    size_t i = (size_t)blockIdx.x * blockDim.x + threadIdx.x;
    if (i < n) out[i] = (_Float16)in[i];
}

__global__ __launch_bounds__(256) void init_ms(float* __restrict__ M,
                                               float* __restrict__ S, int n) {
    int i = blockIdx.x * blockDim.x + threadIdx.x;
    if (i < n) { M[i] = __uint_as_float(0xff800000u); S[i] = 0.0f; }
}

// ---------------------------------------------------------------------------
// WMMA GEMM: Y[16 x 16*NTILES] strip per wave.  Y = X(f16) @ Wᵀ (packed).
// Packed weight (4*NTILES KB) is staged in LDS once per block (4 waves reuse
// it) via async global->LDS loads; B fragments then come from ds_load.
// mode 0: Y -> f16 outH                         (q/k/v projections)
// mode 1: Y -> f32 outF                         (skip branch, inits H buffer)
// mode 2: gelu(Y+b) -> f16 outH                 (MLP hidden)
// mode 3: resid + gelu(Y+b) -> f32 outF AND f16 outH (residual MLP out)
// mode 4: gelu(Y+b) -> f32 outF                 (final output)
// ---------------------------------------------------------------------------
template <int NTILES>
__global__ __launch_bounds__(128) void gemm_node(const _Float16* __restrict__ X,
                                                 const _Float16* __restrict__ Wpk,
                                                 const float* __restrict__ bias,
                                                 const float* __restrict__ resid,
                                                 float* __restrict__ outF,
                                                 _Float16* __restrict__ outH,
                                                 int nRows, int mode) {
    __shared__ __align__(32) _Float16 sW[4 * NTILES * 512];

    // ---- stage packed weights into LDS (16B chunks, whole block) ----
    const int nchunk = 4 * NTILES * 64;              // number of 16B chunks
    for (int c = threadIdx.x; c < nchunk; c += blockDim.x) {
#if HAS_ASYNC_LDS
        __builtin_amdgcn_global_load_async_to_lds_b128(
            (__attribute__((address_space(1))) v4i*)(Wpk + (size_t)c * 8),
            (__attribute__((address_space(3))) v4i*)(sW + (size_t)c * 8),
            0, 0);
#else
        *(v8h*)(sW + (size_t)c * 8) = *(const v8h*)(Wpk + (size_t)c * 8);
#endif
    }
#if HAS_ASYNC_LDS
#  if __has_builtin(__builtin_amdgcn_s_wait_asynccnt)
    __builtin_amdgcn_s_wait_asynccnt(0);
#  else
    asm volatile("s_wait_asynccnt 0x0" ::: "memory");
#  endif
#endif
    __syncthreads();

    const int rowTiles = (nRows + 15) >> 4;
    int wv = blockIdx.x * (blockDim.x >> 5) + (threadIdx.x >> 5);
    if (wv >= rowTiles) return;
    int lane = threadIdx.x & 31;
    int m = lane & 15, half = lane >> 4;
    const int ncols = NTILES * 16;
    int r0 = wv << 4;
    int rA = r0 + m; if (rA >= nRows) rA = nRows - 1;   // clamp (N%16==0 normally)
    const _Float16* xrow = X + (size_t)rA * 128;

    // ---- preload all four A fragments (128B / lane) ----
    AFrag a[4];
#pragma unroll
    for (int kc = 0; kc < 4; ++kc) {
        a[kc].h[0] = *(const v8h*)(xrow + kc * 32 + half * 8);
        a[kc].h[1] = *(const v8h*)(xrow + kc * 32 + 16 + half * 8);
    }

    v8f zero = {};
    v8f acc[NTILES];
#pragma unroll
    for (int t = 0; t < NTILES; ++t) acc[t] = zero;

#pragma unroll
    for (int kc = 0; kc < 4; ++kc) {
#pragma unroll
        for (int nt = 0; nt < NTILES; ++nt) {
            v16h b = *(const v16h*)(sW + (size_t)(kc * NTILES + nt) * 512 + lane * 16);
            acc[nt] = __builtin_amdgcn_wmma_f32_16x16x32_f16(
                false, a[kc].v, false, b, (short)0, acc[nt], false, false);
        }
    }

#pragma unroll
    for (int nt = 0; nt < NTILES; ++nt) {
        int col = nt * 16 + m;
        float bv = bias ? bias[col] : 0.0f;
#pragma unroll
        for (int r = 0; r < 8; ++r) {
            int row = r0 + half * 8 + r;
            if (row >= nRows) continue;
            float v = acc[nt][r] + bv;
            size_t idx = (size_t)row * ncols + col;
            if (mode == 0)      { outH[idx] = (_Float16)v; }
            else if (mode == 1) { outF[idx] = v; }
            else if (mode == 2) { outH[idx] = (_Float16)gelu_tanh(v); }
            else if (mode == 3) { float o = resid[idx] + gelu_tanh(v);
                                  outF[idx] = o; outH[idx] = (_Float16)o; }
            else                { outF[idx] = gelu_tanh(v); }
        }
    }
}

// ---------------------------------------------------------------------------
// Edge pass 1: alpha[e][h] = <q[dst], k[src]+eproj>_h / sqrt(32);  seg-max.
// One wave per edge; lane handles feature h*32+lane for each of the 4 heads.
// ---------------------------------------------------------------------------
__global__ __launch_bounds__(256) void edge_alpha_kernel(
    const _Float16* __restrict__ Q, const _Float16* __restrict__ K,
    const int* __restrict__ ei, const float* __restrict__ ea,
    const float* __restrict__ We, float* __restrict__ alpha,
    float* __restrict__ M, int E) {
    __shared__ float sWe[384];
    for (int i = threadIdx.x; i < 384; i += blockDim.x) sWe[i] = We[i];
    __syncthreads();
    int e = blockIdx.x * (blockDim.x >> 5) + (threadIdx.x >> 5);
    if (e >= E) return;
    int lane = threadIdx.x & 31;
    int src = ei[e], dst = ei[E + e];
    float a0 = ea[(size_t)e * 3], a1 = ea[(size_t)e * 3 + 1], a2 = ea[(size_t)e * 3 + 2];
    float dot[4];
#pragma unroll
    for (int h = 0; h < 4; ++h) {
        int f = h * 32 + lane;
        float ef = a0 * sWe[f * 3] + a1 * sWe[f * 3 + 1] + a2 * sWe[f * 3 + 2];
        float qv = (float)Q[(size_t)dst * 128 + f];
        float kv = (float)K[(size_t)src * 128 + f] + ef;
        dot[h] = qv * kv;
    }
#pragma unroll
    for (int off = 16; off > 0; off >>= 1) {
#pragma unroll
        for (int h = 0; h < 4; ++h) dot[h] += __shfl_xor(dot[h], off, 32);
    }
    if (lane == 0) {
#pragma unroll
        for (int h = 0; h < 4; ++h) {
            float al = dot[h] * 0.17677669529663687f;   // 1/sqrt(32)
            alpha[(size_t)e * 4 + h] = al;
            atomicMaxF(&M[(size_t)dst * 4 + h], al);
        }
    }
}

// Edge pass 2: p = exp(alpha - max); segment-sum (in-place alpha -> p).
__global__ __launch_bounds__(256) void edge_soft_kernel(
    const int* __restrict__ ei, float* __restrict__ alpha,
    const float* __restrict__ M, float* __restrict__ S, int E) {
    int t = blockIdx.x * blockDim.x + threadIdx.x;
    if (t >= E * 4) return;
    int e = t >> 2, h = t & 3;
    int dst = ei[E + e];
    float p = __expf(alpha[t] - M[(size_t)dst * 4 + h]);
    alpha[t] = p;
    atomicAdd(&S[(size_t)dst * 4 + h], p);
}

// Edge pass 3: H[dst] += (p/s) * (v[src] + eproj).  H pre-filled with skip.
__global__ __launch_bounds__(256) void edge_aggr_kernel(
    const _Float16* __restrict__ V, const int* __restrict__ ei,
    const float* __restrict__ ea, const float* __restrict__ We,
    const float* __restrict__ P, const float* __restrict__ S,
    float* __restrict__ H, int E) {
    __shared__ float sWe[384];
    for (int i = threadIdx.x; i < 384; i += blockDim.x) sWe[i] = We[i];
    __syncthreads();
    int e = blockIdx.x * (blockDim.x >> 5) + (threadIdx.x >> 5);
    if (e >= E) return;
    int lane = threadIdx.x & 31;
    int src = ei[e], dst = ei[E + e];
    float a0 = ea[(size_t)e * 3], a1 = ea[(size_t)e * 3 + 1], a2 = ea[(size_t)e * 3 + 2];
#pragma unroll
    for (int h = 0; h < 4; ++h) {
        float a = P[(size_t)e * 4 + h] / fmaxf(S[(size_t)dst * 4 + h], 1e-16f);
        int f = h * 32 + lane;
        float ef = a0 * sWe[f * 3] + a1 * sWe[f * 3 + 1] + a2 * sWe[f * 3 + 2];
        float val = a * ((float)V[(size_t)src * 128 + f] + ef);
        atomicAdd(&H[(size_t)dst * 128 + f], val);
    }
}

// ---------------------------------------------------------------------------
extern "C" void kernel_launch(void* const* d_in, const int* in_sizes, int n_in,
                              void* d_out, int out_size, void* d_ws, size_t ws_size,
                              hipStream_t stream) {
    const float* x   = (const float*)d_in[0];
    const int*   ei  = (const int*)d_in[1];
    const float* ea  = (const float*)d_in[2];
    const float* Wq1 = (const float*)d_in[3];
    const float* Wk1 = (const float*)d_in[4];
    const float* Wv1 = (const float*)d_in[5];
    const float* We1 = (const float*)d_in[6];
    const float* Ws1 = (const float*)d_in[7];
    const float* M1a = (const float*)d_in[8];
    const float* b1a = (const float*)d_in[9];
    const float* M1b = (const float*)d_in[10];
    const float* b1b = (const float*)d_in[11];
    const float* Wq2 = (const float*)d_in[12];
    const float* Wk2 = (const float*)d_in[13];
    const float* Wv2 = (const float*)d_in[14];
    const float* We2 = (const float*)d_in[15];
    const float* Ws2 = (const float*)d_in[16];
    const float* M2a = (const float*)d_in[17];
    const float* b2a = (const float*)d_in[18];
    const float* M2b = (const float*)d_in[19];
    const float* b2b = (const float*)d_in[20];
    const float* Wf1 = (const float*)d_in[21];
    const float* bf1 = (const float*)d_in[22];
    const float* Wf2 = (const float*)d_in[23];
    const float* bf2 = (const float*)d_in[24];

    const int N = in_sizes[0] / 128;
    const int E = in_sizes[2] / 3;
    const int rowTiles = (N + 15) >> 4;

    // ---- workspace carve-up (256B aligned) ----
    char* ws = (char*)d_ws;
    size_t off = 0;
    auto alloc = [&](size_t bytes) -> char* {
        char* p = ws + off;
        off = (off + bytes + 255) & ~(size_t)255;
        return p;
    };
    _Float16* X16   = (_Float16*)alloc((size_t)N * 128 * 2);
    _Float16* T16   = (_Float16*)alloc((size_t)N * 128 * 2);
    _Float16* Qb    = (_Float16*)alloc((size_t)N * 128 * 2);
    _Float16* Kb    = (_Float16*)alloc((size_t)N * 128 * 2);
    _Float16* Vb    = (_Float16*)alloc((size_t)N * 128 * 2);
    float*    H0    = (float*)alloc((size_t)N * 128 * 4);
    float*    H1    = (float*)alloc((size_t)N * 128 * 4);
    float*    ALPHA = (float*)alloc((size_t)E * 4 * 4);
    float*    Mb    = (float*)alloc((size_t)N * 4 * 4);
    float*    Sb    = (float*)alloc((size_t)N * 4 * 4);

    const float* wsrc[14] = {Wq1, Wk1, Wv1, Ws1, M1a, M1b,
                             Wq2, Wk2, Wv2, Ws2, M2a, M2b, Wf1, Wf2};
    _Float16* pw[14];
    for (int i = 0; i < 14; ++i) {
        int ntiles = (i == 13) ? 4 : 8;
        pw[i] = (_Float16*)alloc((size_t)4 * ntiles * 512 * 2);
    }
    // pack weights into WMMA B-fragment layout
    for (int i = 0; i < 14; ++i) {
        int ntiles = (i == 13) ? 4 : 8;
        pack_w<<<dim3(4 * ntiles), dim3(32), 0, stream>>>(wsrc[i], pw[i], ntiles);
    }

    const int gemmBlocks = (rowTiles + 3) / 4;     // 4 waves / block
    const int convBlocks = (int)(((size_t)N * 128 + 255) / 256);
    const int edgeBlocks = (E + 7) / 8;            // 8 edges / block (wave each)
    const int msBlocks   = (N * 4 + 255) / 256;
    const int softBlocks = (E * 4 + 255) / 256;

    auto G = [&](const _Float16* Xin, _Float16* Wp, const float* b,
                 const float* resid, float* oF, _Float16* oH, int mode) {
        gemm_node<8><<<dim3(gemmBlocks), dim3(128), 0, stream>>>(
            Xin, Wp, b, resid, oF, oH, N, mode);
    };

    // ---------------- layer 1 ----------------
    conv16<<<dim3(convBlocks), dim3(256), 0, stream>>>(x, X16, (size_t)N * 128);
    G(X16, pw[0], nullptr, nullptr, nullptr, Qb, 0);
    G(X16, pw[1], nullptr, nullptr, nullptr, Kb, 0);
    G(X16, pw[2], nullptr, nullptr, nullptr, Vb, 0);
    G(X16, pw[3], nullptr, nullptr, H0, nullptr, 1);          // skip -> H0
    init_ms<<<dim3(msBlocks), dim3(256), 0, stream>>>(Mb, Sb, N * 4);
    edge_alpha_kernel<<<dim3(edgeBlocks), dim3(256), 0, stream>>>(Qb, Kb, ei, ea, We1, ALPHA, Mb, E);
    edge_soft_kernel<<<dim3(softBlocks), dim3(256), 0, stream>>>(ei, ALPHA, Mb, Sb, E);
    edge_aggr_kernel<<<dim3(edgeBlocks), dim3(256), 0, stream>>>(Vb, ei, ea, We1, ALPHA, Sb, H0, E);
    // residual MLP 1
    conv16<<<dim3(convBlocks), dim3(256), 0, stream>>>(H0, X16, (size_t)N * 128);
    G(X16, pw[4], b1a, nullptr, nullptr, T16, 2);
    G(T16, pw[5], b1b, H0, H0, X16, 3);                       // H0 += gelu(...), X16 = f16(H0)

    // ---------------- layer 2 ----------------
    G(X16, pw[6], nullptr, nullptr, nullptr, Qb, 0);
    G(X16, pw[7], nullptr, nullptr, nullptr, Kb, 0);
    G(X16, pw[8], nullptr, nullptr, nullptr, Vb, 0);
    G(X16, pw[9], nullptr, nullptr, H1, nullptr, 1);          // skip -> H1
    init_ms<<<dim3(msBlocks), dim3(256), 0, stream>>>(Mb, Sb, N * 4);
    edge_alpha_kernel<<<dim3(edgeBlocks), dim3(256), 0, stream>>>(Qb, Kb, ei, ea, We2, ALPHA, Mb, E);
    edge_soft_kernel<<<dim3(softBlocks), dim3(256), 0, stream>>>(ei, ALPHA, Mb, Sb, E);
    edge_aggr_kernel<<<dim3(edgeBlocks), dim3(256), 0, stream>>>(Vb, ei, ea, We2, ALPHA, Sb, H1, E);
    // residual MLP 2
    conv16<<<dim3(convBlocks), dim3(256), 0, stream>>>(H1, X16, (size_t)N * 128);
    G(X16, pw[10], b2a, nullptr, nullptr, T16, 2);
    G(T16, pw[11], b2b, H1, H1, X16, 3);                      // H1 += gelu(...), X16 = f16(H1)

    // ---------------- final MLP ----------------
    G(X16, pw[12], bf1, nullptr, nullptr, T16, 2);
    gemm_node<4><<<dim3(gemmBlocks), dim3(128), 0, stream>>>(
        T16, pw[13], bf2, nullptr, (float*)d_out, nullptr, N, 4);
}
